// MaskedCrossAttention_12945031430319
// MI455X (gfx1250) — matrix operand order repaired and checked by reading, not compiled, not run on hardware
//
#include <hip/hip_runtime.h>
#include <hip/hip_bf16.h>
#include <stdint.h>

// ---------------------------------------------------------------------------
// MaskedCrossAttention on MI455X (gfx1250, wave32, WMMA + async-LDS + TDM).
// bf16 WMMA inputs, f32 accumulate (reference is f32).
// ---------------------------------------------------------------------------

#ifndef __has_builtin
#define __has_builtin(x) 0
#endif

typedef __bf16 bf16;
typedef __attribute__((ext_vector_type(16))) __bf16 v16bf;
typedef __attribute__((ext_vector_type(8)))  __bf16 v8bf;
typedef __attribute__((ext_vector_type(8)))  float  v8f;
typedef __attribute__((ext_vector_type(4)))  int    v4i;
typedef __attribute__((ext_vector_type(4)))  unsigned int uint32x4;
typedef __attribute__((ext_vector_type(8)))  int int32x8;
typedef __attribute__((ext_vector_type(4)))  int int32x4;

#define B_    4
#define T_TXT 2048
#define DIM_  2048
#define DVIS  1024
#define HEADS 16
#define DHEAD 64
#define INNER 1024
#define NKEY  512
#define SCALE_ 0.125f

#if __has_builtin(__builtin_amdgcn_tensor_load_to_lds)
#define HAVE_TDM 1
#else
#define HAVE_TDM 0
#endif

__device__ __forceinline__ bf16 f2bf(float f) { return (bf16)f; }

__device__ __forceinline__ v16bf cat16(v8bf lo, v8bf hi) {
  return __builtin_shufflevector(lo, hi, 0,1,2,3,4,5,6,7,8,9,10,11,12,13,14,15);
}

// A-operand (16x32, row-major, stride lda elems)
__device__ __forceinline__ v16bf load_A(const bf16* base, int lda, int lane) {
  int m  = lane & 15;
  int kb = (lane < 16) ? 0 : 8;
  const bf16* p = base + (size_t)m * lda + kb;
  return cat16(*(const v8bf*)(p), *(const v8bf*)(p + 16));
}

// B-operand (32x16) from N-major storage BT[n][k], stride ldb elems
__device__ __forceinline__ v16bf load_B(const bf16* base, int ldb, int lane) {
  int n  = lane & 15;
  int kb = (lane < 16) ? 0 : 8;
  const bf16* p = base + (size_t)n * ldb + kb;
  return cat16(*(const v8bf*)(p), *(const v8bf*)(p + 16));
}

__device__ __forceinline__ v8f wmma_bf16(v16bf a, v16bf b, v8f c) {
  return __builtin_amdgcn_wmma_f32_16x16x32_bf16(false, a, false, b,
                                                 (short)0, c, false, false);
}

// --- CDNA5 async global->LDS copy (16B per lane), ASYNCcnt-tracked ---------
__device__ __forceinline__ void async_g2l_b128(const bf16* gptr, bf16* lptr) {
#if __has_builtin(__builtin_amdgcn_global_load_async_to_lds_b128)
  __builtin_amdgcn_global_load_async_to_lds_b128((v4i*)gptr, (v4i*)lptr, 0, 0);
#else
  uint32_t l = (uint32_t)(uintptr_t)lptr;   // low 32 bits of flat LDS addr = LDS offset
  asm volatile("global_load_async_to_lds_b128 %0, %1, off"
               :: "v"(l), "v"(gptr) : "memory");
#endif
}

__device__ __forceinline__ void wait_async1() {
#if __has_builtin(__builtin_amdgcn_s_wait_asynccnt)
  __builtin_amdgcn_s_wait_asynccnt(1);
#else
  asm volatile("s_wait_asynccnt 0x1" ::: "memory");
#endif
}
__device__ __forceinline__ void wait_async0() {
#if __has_builtin(__builtin_amdgcn_s_wait_asynccnt)
  __builtin_amdgcn_s_wait_asynccnt(0);
#else
  asm volatile("s_wait_asynccnt 0x0" ::: "memory");
#endif
}

// --- TDM: 2D tile (nrows x row8*8 bytes) global -> LDS, TENSORcnt-tracked --
#if HAVE_TDM
__device__ __forceinline__ void tdm_load_2d(const void* gaddr, uint32_t lds_addr,
                                            uint32_t row8, uint32_t nrows,
                                            uint32_t stride8) {
  uint64_t ga = (uint64_t)(uintptr_t)gaddr;
  uint32x4 g0;
  g0.x = 1u;                                    // count=1, user descriptor
  g0.y = lds_addr;                              // lds_addr (bytes)
  g0.z = (uint32_t)ga;                          // global_addr[31:0]
  g0.w = (uint32_t)(ga >> 32) | (2u << 30);     // global_addr[56:32] | type=2
  int32x8 g1;
  g1[0] = (int)(3u << 16);                      // data_size = 8B, no multicast
  g1[1] = (int)((stride8 & 0xFFFFu) << 16);     // tensor_dim0[15:0]
  g1[2] = (int)((stride8 >> 16) | ((nrows & 0xFFFFu) << 16));  // td0 hi | td1 lo
  g1[3] = (int)((nrows >> 16) | (row8 << 16));  // td1 hi | tile_dim0
  g1[4] = (int)(nrows & 0xFFFFu);               // tile_dim1 (tile_dim2 = 0)
  g1[5] = (int)stride8;                         // tensor_dim0_stride[31:0]
  g1[6] = 0;
  g1[7] = 0;
  int32x4 z4 = {0, 0, 0, 0};
  int32x8 z8 = {0, 0, 0, 0, 0, 0, 0, 0};
  __builtin_amdgcn_tensor_load_to_lds(g0, g1, z4, z4, z8, 0);
}
#endif

// ---------------------------------------------------------------------------
// Prep kernels
// ---------------------------------------------------------------------------
__global__ void cvtT_kernel(const float* __restrict__ W, bf16* __restrict__ WT,
                            int K, int N) {
  size_t idx = (size_t)blockIdx.x * 256 + threadIdx.x;
  if (idx >= (size_t)K * N) return;
  int n = (int)(idx / K), k = (int)(idx % K);
  WT[idx] = f2bf(W[(size_t)k * N + n]);
}

__global__ void cvt_kernel(const float* __restrict__ in, bf16* __restrict__ out,
                           size_t n) {
  size_t idx = (size_t)blockIdx.x * 256 + threadIdx.x;
  if (idx < n) out[idx] = f2bf(in[idx]);
}

__global__ __launch_bounds__(256)
void ln_kernel(const float* __restrict__ x, const float* __restrict__ w,
               const float* __restrict__ bia, bf16* __restrict__ out) {
  const int D = DIM_;
  int row = blockIdx.x;
  const float* xr = x + (size_t)row * D;
  float s = 0.f, ss = 0.f;
  for (int c = threadIdx.x; c < D; c += 256) { float v = xr[c]; s += v; ss += v * v; }
  __shared__ float red0[256], red1[256];
  red0[threadIdx.x] = s; red1[threadIdx.x] = ss;
  __syncthreads();
  for (int off = 128; off > 0; off >>= 1) {
    if ((int)threadIdx.x < off) {
      red0[threadIdx.x] += red0[threadIdx.x + off];
      red1[threadIdx.x] += red1[threadIdx.x + off];
    }
    __syncthreads();
  }
  float mu  = red0[0] / D;
  float var = red1[0] / D - mu * mu;
  float rstd = rsqrtf(var + 1e-5f);
  bf16* orow = out + (size_t)row * D;
  for (int c = threadIdx.x; c < D; c += 256)
    orow[c] = f2bf((xr[c] - mu) * rstd * w[c] + bia[c]);
}

__global__ void tt_kernel(const unsigned char* __restrict__ loc,
                          int* __restrict__ tt) {
  __shared__ int flags[T_TXT];
  int b = blockIdx.x;
  for (int t = threadIdx.x; t < T_TXT; t += blockDim.x)
    flags[t] = loc[(size_t)b * T_TXT + t] ? 1 : 0;
  __syncthreads();
  if (threadIdx.x == 0) {
    int acc = 0;
    for (int t = 0; t < T_TXT; ++t) { acc += flags[t]; flags[t] = acc; }
  }
  __syncthreads();
  for (int t = threadIdx.x; t < T_TXT; t += blockDim.x)
    tt[(size_t)b * T_TXT + t] = flags[t];
}

// ---------------------------------------------------------------------------
// GEMM: C[M,N] = A[M,K] * B (given N-major as BT[N,K]).
// Block = 8 waves; wave tile 32x64 (8 accum tiles); block tile 256x64.
// B tile (64x32 = 4KB) staged to LDS via async-to-LDS, double buffered.
// ---------------------------------------------------------------------------
#define GEMM_BK 32
template <typename TO>
__global__ __launch_bounds__(256)
void gemm_bf16(const bf16* __restrict__ A, const bf16* __restrict__ BT,
               TO* __restrict__ C, int M, int N, int K) {
  __shared__ __attribute__((aligned(128))) bf16 bstage[2][64 * GEMM_BK];
  int lane = threadIdx.x & 31;
  int wave = threadIdx.x >> 5;
  int tid  = threadIdx.x;
  int mo = blockIdx.y * 256 + wave * 32;
  int no = blockIdx.x * 64;

  // staging assignment: 256 threads x 16B = one 4KB tile per stage
  int sn = tid >> 2;                   // staging row (0..63)
  int sq = tid & 3;                    // 16B segment within row
  const bf16* bsrc = BT + (size_t)(no + sn) * K + sq * 8;
  bf16* bdst0 = &bstage[0][sn * GEMM_BK + sq * 8];
  bf16* bdst1 = &bstage[1][sn * GEMM_BK + sq * 8];

  int nIter = K / GEMM_BK;
  async_g2l_b128(bsrc, bdst0);                       // stage 0

  v8f acc[2][4] = {};
  for (int it = 0; it < nIter; ++it) {
    int k0 = it * GEMM_BK;
    if (it + 1 < nIter) {
      async_g2l_b128(bsrc + k0 + GEMM_BK, (it & 1) ? bdst0 : bdst1);
      wait_async1();                                 // current stage landed
    } else {
      wait_async0();
    }
    __syncthreads();                                 // all waves' portions in LDS
    const bf16* bs = &bstage[it & 1][0];
    v16bf av0 = load_A(A + (size_t)mo * K + k0, K, lane);
    v16bf av1 = load_A(A + (size_t)(mo + 16) * K + k0, K, lane);
#pragma unroll
    for (int nt = 0; nt < 4; ++nt) {
      v16bf bv = load_B(bs + nt * 16 * GEMM_BK, GEMM_BK, lane);
      acc[0][nt] = wmma_bf16(av0, bv, acc[0][nt]);
      acc[1][nt] = wmma_bf16(av1, bv, acc[1][nt]);
    }
    __syncthreads();                                 // done reading this buffer
  }

  int half = (lane < 16) ? 0 : 8;
  int ncol = lane & 15;
#pragma unroll
  for (int mt = 0; mt < 2; ++mt)
#pragma unroll
    for (int nt = 0; nt < 4; ++nt)
#pragma unroll
      for (int r = 0; r < 8; ++r) {
        size_t row = (size_t)(mo + mt * 16 + r + half);
        int col = no + nt * 16 + ncol;
        float v = acc[mt][nt][r];
        if constexpr (sizeof(TO) == 4) C[row * N + col] = v;
        else                           C[row * N + col] = f2bf(v);
      }
}

// Repack kv GEMM output into per-head K[b][h][key][d] and VT[b][h][d][key]
__global__ void repack_kv(const bf16* __restrict__ kvb, bf16* __restrict__ kst,
                          bf16* __restrict__ vT) {
  size_t idx = (size_t)blockIdx.x * 256 + threadIdx.x;
  if (idx >= (size_t)(B_ * NKEY) * (2 * INNER)) return;
  int row = (int)(idx >> 11);
  int col = (int)(idx & 2047);
  int b = row >> 9, key = row & 511;
  bf16 val = kvb[idx];
  if (col < INNER) {
    int h = col >> 6, d = col & 63;
    kst[(((size_t)(b * HEADS + h) * NKEY) + key) * DHEAD + d] = val;
  } else {
    int c2 = col - INNER;
    int h = c2 >> 6, d = c2 & 63;
    vT[(((size_t)(b * HEADS + h) * DHEAD) + d) * NKEY + key] = val;
  }
}

// ---------------------------------------------------------------------------
// Attention: 4 waves/block, one wave per (b,h,16-query tile).
// K/V chunk tiles fetched with the Tensor Data Mover into per-wave LDS.
// ---------------------------------------------------------------------------
__global__ __launch_bounds__(128)
void attn_kernel(const bf16* __restrict__ qb, const bf16* __restrict__ kst,
                 const bf16* __restrict__ vT, const int* __restrict__ tt,
                 bf16* __restrict__ aout) {
  __shared__ __attribute__((aligned(128))) bf16 kt_s[4][32 * 64];   // 16KB
  __shared__ __attribute__((aligned(128))) bf16 vt_s[4][64 * 32];   // 16KB
  __shared__ __attribute__((aligned(64)))  bf16 pbuf_all[4][16 * 32]; // 4KB
  int lane = threadIdx.x & 31;
  int wave = threadIdx.x >> 5;
  bf16* pbuf = &pbuf_all[wave][0];
  bf16* kt   = &kt_s[wave][0];
  bf16* vt   = &vt_s[wave][0];

  int job = blockIdx.x * 4 + wave;           // 8192 jobs
  int qt = job & 127;
  int rest = job >> 7;
  int h = rest & 15;
  int b = rest >> 4;
  int qbase = qt * 16;

  const bf16* qrow = qb + ((size_t)(b * T_TXT + qbase) + (lane & 15)) * INNER + h * DHEAD;
  int kb = (lane < 16) ? 0 : 8;
  v16bf qa0 = cat16(*(const v8bf*)(qrow + kb),      *(const v8bf*)(qrow + kb + 16));
  v16bf qa1 = cat16(*(const v8bf*)(qrow + 32 + kb), *(const v8bf*)(qrow + 32 + kb + 16));

  int half = (lane < 16) ? 0 : 8;
  int ttl[8];
#pragma unroll
  for (int r = 0; r < 8; ++r)
    ttl[r] = tt[(size_t)b * T_TXT + qbase + r + half];

  const bf16* kbase = kst + (size_t)(b * HEADS + h) * NKEY * DHEAD;
  const bf16* vbase = vT  + (size_t)(b * HEADS + h) * DHEAD * NKEY;

  float rmax[8], rsum[8];
  unsigned hasv = 0;
#pragma unroll
  for (int r = 0; r < 8; ++r) { rmax[r] = -1e30f; rsum[r] = 0.f; }
  v8f o[4] = {};

  for (int c = 0; c < NKEY / 32; ++c) {
    int j = (c >> 1) + 1;
    bool anyv = false;
#pragma unroll
    for (int r = 0; r < 8; ++r) anyv |= (ttl[r] == j);
    if (!__any(anyv)) continue;

    v8f s0 = {}, s1 = {};
#if HAVE_TDM
    // K tile: 32 keys x 64 feats (128B rows); V tile: 64 feats x 32 keys (64B rows)
    tdm_load_2d(kbase + (size_t)(c * 32) * DHEAD,
                (uint32_t)(uintptr_t)kt, 16u, 32u, (uint32_t)(DHEAD / 4));
    tdm_load_2d(vbase + c * 32,
                (uint32_t)(uintptr_t)vt, 8u, 64u, (uint32_t)(NKEY / 4));
    __builtin_amdgcn_s_wait_tensorcnt(0);
    s0 = wmma_bf16(qa0, load_B(kt, DHEAD, lane), s0);
    s0 = wmma_bf16(qa1, load_B(kt + 32, DHEAD, lane), s0);
    s1 = wmma_bf16(qa0, load_B(kt + 16 * DHEAD, DHEAD, lane), s1);
    s1 = wmma_bf16(qa1, load_B(kt + 16 * DHEAD + 32, DHEAD, lane), s1);
#else
    const bf16* k0p = kbase + (size_t)(c * 32) * DHEAD;
    s0 = wmma_bf16(qa0, load_B(k0p, DHEAD, lane), s0);
    s0 = wmma_bf16(qa1, load_B(k0p + 32, DHEAD, lane), s0);
    s1 = wmma_bf16(qa0, load_B(k0p + 16 * DHEAD, DHEAD, lane), s1);
    s1 = wmma_bf16(qa1, load_B(k0p + 16 * DHEAD + 32, DHEAD, lane), s1);
#endif

    float a_r[8];
    int ncol = lane & 15;
#pragma unroll
    for (int r = 0; r < 8; ++r) {
      bool valid = (ttl[r] == j);
      if (valid) hasv |= (1u << r);
      float v0 = valid ? s0[r] * SCALE_ : -1e30f;
      float v1 = valid ? s1[r] * SCALE_ : -1e30f;
      float cm = fmaxf(v0, v1);
#pragma unroll
      for (int off = 1; off < 16; off <<= 1)
        cm = fmaxf(cm, __shfl_xor(cm, off, 32));
      float nm = fmaxf(rmax[r], cm);
      float alpha = __expf(rmax[r] - nm);
      float p0 = __expf(v0 - nm);
      float p1 = __expf(v1 - nm);
      float ps = p0 + p1;
#pragma unroll
      for (int off = 1; off < 16; off <<= 1)
        ps += __shfl_xor(ps, off, 32);
      rsum[r] = rsum[r] * alpha + ps;
      rmax[r] = nm;
      a_r[r] = alpha;
      int m = r + half;
      pbuf[m * 32 + ncol]      = f2bf(p0);
      pbuf[m * 32 + 16 + ncol] = f2bf(p1);
    }
#pragma unroll
    for (int nt = 0; nt < 4; ++nt)
#pragma unroll
      for (int r = 0; r < 8; ++r) o[nt][r] *= a_r[r];

    asm volatile("s_wait_dscnt 0" ::: "memory");

    v16bf pa = load_A(pbuf, 32, lane);
#pragma unroll
    for (int nt = 0; nt < 4; ++nt) {
#if HAVE_TDM
      v16bf bv = load_B(vt + (size_t)(nt * 16) * 32, 32, lane);
#else
      v16bf bv = load_B(vbase + (size_t)(nt * 16) * NKEY + c * 32, NKEY, lane);
#endif
      o[nt] = wmma_bf16(pa, bv, o[nt]);
    }
  }

  float sc[8];
#pragma unroll
  for (int r = 0; r < 8; ++r)
    sc[r] = (hasv & (1u << r)) ? 1.0f / rsum[r] : 0.f;
  size_t obase = (size_t)(b * T_TXT + qbase) * INNER + h * DHEAD;
  int ncol = lane & 15;
#pragma unroll
  for (int nt = 0; nt < 4; ++nt)
#pragma unroll
    for (int r = 0; r < 8; ++r)
      aout[obase + (size_t)(r + half) * INNER + nt * 16 + ncol] =
          f2bf(o[nt][r] * sc[r]);
}

// ---------------------------------------------------------------------------
// Host launch
// ---------------------------------------------------------------------------
extern "C" void kernel_launch(void* const* d_in, const int* in_sizes, int n_in,
                              void* d_out, int out_size, void* d_ws, size_t ws_size,
                              hipStream_t stream) {
  const float* x     = (const float*)d_in[0];
  const float* media = (const float*)d_in[1];
  const unsigned char* locs = (const unsigned char*)d_in[2];
  const float* ln_w  = (const float*)d_in[3];
  const float* ln_b  = (const float*)d_in[4];
  const float* Wq    = (const float*)d_in[5];
  const float* Wkv   = (const float*)d_in[6];
  const float* Wo    = (const float*)d_in[7];
  float* out = (float*)d_out;

  char* w = (char*)d_ws;
  auto take = [&](size_t bytes) {
    char* p = w;
    w += (bytes + 255) & ~(size_t)255;
    return p;
  };
  bf16* xnb  = (bf16*)take((size_t)B_ * T_TXT * DIM_ * 2);
  bf16* qbuf = (bf16*)take((size_t)B_ * T_TXT * INNER * 2);
  bf16* wqT  = (bf16*)take((size_t)DIM_ * INNER * 2);
  bf16* wkvT = (bf16*)take((size_t)DVIS * 2 * INNER * 2);
  bf16* woT  = (bf16*)take((size_t)INNER * DIM_ * 2);
  bf16* medb = (bf16*)take((size_t)B_ * NKEY * DVIS * 2);
  bf16* kvb  = (bf16*)take((size_t)B_ * NKEY * 2 * INNER * 2);
  bf16* kstp = (bf16*)take((size_t)B_ * HEADS * NKEY * DHEAD * 2);
  bf16* vTp  = (bf16*)take((size_t)B_ * HEADS * DHEAD * NKEY * 2);
  bf16* aout = (bf16*)take((size_t)B_ * T_TXT * INNER * 2);
  int*  ttp  = (int*)take((size_t)B_ * T_TXT * 4);

  {
    size_t n = (size_t)DIM_ * INNER;
    cvtT_kernel<<<(unsigned)((n + 255) / 256), 256, 0, stream>>>(Wq, wqT, DIM_, INNER);
  }
  {
    size_t n = (size_t)DVIS * 2 * INNER;
    cvtT_kernel<<<(unsigned)((n + 255) / 256), 256, 0, stream>>>(Wkv, wkvT, DVIS, 2 * INNER);
  }
  {
    size_t n = (size_t)INNER * DIM_;
    cvtT_kernel<<<(unsigned)((n + 255) / 256), 256, 0, stream>>>(Wo, woT, INNER, DIM_);
  }
  {
    size_t n = (size_t)B_ * NKEY * DVIS;
    cvt_kernel<<<(unsigned)((n + 255) / 256), 256, 0, stream>>>(media, medb, n);
  }

  ln_kernel<<<B_ * T_TXT, 256, 0, stream>>>(x, ln_w, ln_b, xnb);
  tt_kernel<<<B_, 256, 0, stream>>>(locs, ttp);

  gemm_bf16<bf16><<<dim3(INNER / 64, (B_ * T_TXT) / 256), 256, 0, stream>>>(
      xnb, wqT, qbuf, B_ * T_TXT, INNER, DIM_);

  gemm_bf16<bf16><<<dim3((2 * INNER) / 64, (B_ * NKEY) / 256), 256, 0, stream>>>(
      medb, wkvT, kvb, B_ * NKEY, 2 * INNER, DVIS);
  {
    size_t n = (size_t)B_ * NKEY * 2 * INNER;
    repack_kv<<<(unsigned)((n + 255) / 256), 256, 0, stream>>>(kvb, kstp, vTp);
  }

  attn_kernel<<<(B_ * HEADS * (T_TXT / 16)) / 4, 128, 0, stream>>>(
      qbuf, kstp, vTp, ttp, aout);

  gemm_bf16<float><<<dim3(DIM_ / 64, (B_ * T_TXT) / 256), 256, 0, stream>>>(
      aout, woT, out, B_ * T_TXT, DIM_, INNER);
}
